// AttentionModule_24068996726850
// MI455X (gfx1250) — compile-verified, matrix operand order checked by reading, not run on hardware
//
#include <hip/hip_runtime.h>

// ---------------------------------------------------------------------------
// Fused AttentionModule for MI455X (gfx1250, wave32, WMMA + pipelined TDM).
//
// Exact simplification: the L=16 expand makes every L-column identical, so the
// softmax is exactly uniform and the attention path cancels:
//   out = relu(bn2(Wv2 @ relu(bn1(Wv1 @ f2)))) + pos * sigmoid(Wvc @ pos),
//   pos = relu(Wk @ f).
// Four N x 512 x 512 bf16 WMMA GEMMs fused per 64-row slab; weights are
// pre-converted to bf16 once (prep kernel). Weight K-tiles are DMA'd into a
// double-buffered LDS region by the Tensor Data Mover (pad_enable reproduces
// the padded LDS layout), overlapped with WMMA compute via split
// s_wait_tensorcnt (<=1 while a tile is still in flight).
// ---------------------------------------------------------------------------

typedef __attribute__((ext_vector_type(16))) __bf16         v16bf;
typedef __attribute__((ext_vector_type(8)))  float          v8f;
typedef __attribute__((ext_vector_type(4)))  unsigned int   u32x4;
typedef __attribute__((ext_vector_type(8)))  int            i32x8;
typedef __attribute__((ext_vector_type(4)))  int            i32x4;
typedef __attribute__((ext_vector_type(4)))  float          f32x4;
typedef __attribute__((ext_vector_type(4)))  unsigned short u16x4;

#define NROIS   65536
#define CDIM    512
#define DDIM    512
#define MT      64          // rows per workgroup
#define XSTR    520         // bf16 elems per LDS input row (512 + 8 pad)
#define WSTR    40          // bf16 elems per W-tile row   (32 + 8 pad)
#define KSTEP   32
#define BN_EPS  1e-5f
#define WELEMS  (CDIM * DDIM)       // 262144 elems per weight matrix
#define WTILE_E (DDIM * WSTR)       // 20480 ushorts per W-tile buffer

// LDS byte offsets
#define LDS_BUF0  0                       // features tile, reused for h
#define LDS_BUF1  66560                   // features2 tile  (64*520*2)
#define LDS_POS   133120                  // pos tile
#define LDS_W0    199680                  // W K-tile buffer 0: 512 x WSTR bf16
#define LDS_W1    240640                  // W K-tile buffer 1
#define LDS_TOTAL 281600                  // 240640 + 40960

#if __has_builtin(__builtin_amdgcn_tensor_load_to_lds) && \
    __has_builtin(__builtin_amdgcn_s_wait_tensorcnt)
#define USE_TDM 1
#else
#define USE_TDM 0
#endif

union Frag16 { v16bf v; u32x4 q[2]; unsigned short s[16]; };

__device__ __forceinline__ unsigned short f2bf(float f) {
    return __builtin_bit_cast(unsigned short, (__bf16)f);   // native v_cvt
}
__device__ __forceinline__ float bf2f(unsigned short s) {
    unsigned u = ((unsigned)s) << 16;
    return __builtin_bit_cast(float, u);
}

#if USE_TDM
// Tensor DMA: load W[0..511][k0..k0+31] (bf16, row stride 512) into LDS at
// byte offset ldsOff with a 16B pad after every 64B row -> [512][WSTR] tile.
// D# per CDNA5 ISA 8.3/8.4; groups 2/3 zero (2D tensor). 6-arg builtin.
__device__ __forceinline__ void tdm_load_wtile(const unsigned short* wsrc,
                                               int k0, unsigned ldsOff) {
    unsigned long long ga =
        (unsigned long long)(const void*)wsrc + (unsigned long long)(k0 * 2);
    u32x4 g0;
    g0.x = 1u;                                   // count=1, user descriptor
    g0.y = ldsOff;                               // lds_addr (bytes)
    g0.z = (unsigned)ga;                         // global_addr[31:0]
    g0.w = (unsigned)((ga >> 32) & 0x01FFFFFFull) | (2u << 30);  // [56:32]|type=2
    i32x8 g1;
    g1[0] = (1 << 16)      // data_size = 2B
          | (1 << 20)      // pad_enable
          | (3 << 22)      // pad_interval: 16 DWORDs (64B = one 32-elem row)
          | (3 << 25);     // pad_amount:    4 DWORDs (16B = 8 bf16)
    g1[1] = (CDIM & 0xFFFF) << 16;               // tensor_dim0[15:0]
    g1[2] = (DDIM & 0xFFFF) << 16;               // td0 hi=0 | tensor_dim1[15:0]
    g1[3] = (KSTEP & 0xFFFF) << 16;              // td1 hi=0 | tile_dim0 = 32
    g1[4] = DDIM;                                // tile_dim1 = 512 | tile_dim2=0
    g1[5] = CDIM;                                // tensor_dim0_stride[31:0]
    g1[6] = 0;
    g1[7] = 0;
    i32x4 z4 = (i32x4){0, 0, 0, 0};
    i32x8 z8 = (i32x8){0, 0, 0, 0, 0, 0, 0, 0};
    __builtin_amdgcn_tensor_load_to_lds(g0, g1, z4, z4, z8, 0);
}
#endif

// EPI: 0 = relu -> sDst(bf16)        1 = bn+relu -> sDst(bf16)
//      2 = bn+relu -> accOut(regs)   3 = raw -> accOut(regs)
template<int EPI>
__device__ __forceinline__ void gemm_pass(
    const unsigned short* __restrict__ sX,   // [MT][XSTR] bf16, full K=512
    unsigned short*       __restrict__ sW,   // 2 x [512][WSTR] staging (LDS_W0)
    const unsigned short* __restrict__ Wg,   // global weights [512][512] bf16
    unsigned short*       __restrict__ sDst,
    v8f*                  accOut,
    const float* __restrict__ gm, const float* __restrict__ bt,
    const float* __restrict__ mu, const float* __restrict__ vr,
    int tid)
{
    const int lane  = tid & 31;
    const int wave  = tid >> 5;
    const int mi    = wave & 3;          // 16-row tile index (0..3)
    const int nb0   = (wave >> 2) * 8;   // first of 8 col-tiles
    const int lhalf = lane >> 4;
    const int l15   = lane & 15;

    v8f acc[8];
    #pragma unroll
    for (int j = 0; j < 8; ++j)
        acc[j] = (v8f){0.f,0.f,0.f,0.f,0.f,0.f,0.f,0.f};

    const int am  = mi * 16 + l15;       // A-matrix row for this lane
    const int akb = lhalf * 8;           // K sub-offset per ISA 16-bit layout

#if USE_TDM
    if (wave == 0) {                     // pipeline prologue: tiles 0 and 1
        tdm_load_wtile(Wg, 0,     LDS_W0);
        tdm_load_wtile(Wg, KSTEP, LDS_W1);
    }
#endif

    for (int k0 = 0; k0 < CDIM; k0 += KSTEP) {
        const int cur = (k0 >> 5) & 1;   // parity buffer for tile k
        unsigned short* sWc = sW + cur * WTILE_E;
#if USE_TDM
        if (wave == 0) {                 // tile k resident; k+1 may be in flight
            if (k0 + KSTEP < CDIM) __builtin_amdgcn_s_wait_tensorcnt((short)1);
            else                   __builtin_amdgcn_s_wait_tensorcnt((short)0);
        }
        __syncthreads();                 // publish tile k to all waves
#else
        __syncthreads();                 // previous users of sWc are done
        {   // fallback: cooperative bf16 copy; thread t owns row o = t
            const unsigned short* wrow = Wg + (size_t)tid * CDIM + k0;
            unsigned short* dst = sWc + tid * WSTR;
            #pragma unroll
            for (int i = 0; i < 4; ++i)
                *(u32x4*)(dst + 8 * i) = *(const u32x4*)(wrow + 8 * i);
        }
        __syncthreads();
#endif

        // A fragment (16x32 bf16, ISA 7.12.2 layout): two 16B LDS reads
        Frag16 af;
        const unsigned short* arow = sX + am * XSTR + k0 + akb;
        af.q[0] = *(const u32x4*)(arow);        // K = akb .. akb+7
        af.q[1] = *(const u32x4*)(arow + 16);   // K = akb+16 .. akb+23

        #pragma unroll
        for (int j = 0; j < 8; ++j) {
            Frag16 bfr;                          // column o fixed per lane
            const unsigned short* brow =
                sWc + ((nb0 + j) * 16 + l15) * WSTR + akb;
            bfr.q[0] = *(const u32x4*)(brow);
            bfr.q[1] = *(const u32x4*)(brow + 16);
            acc[j] = __builtin_amdgcn_wmma_f32_16x16x32_bf16(
                false, af.v, false, bfr.v, (short)0, acc[j], false, false);
        }

        __syncthreads();                 // all waves done reading buffer cur
#if USE_TDM
        if (wave == 0 && k0 + 2 * KSTEP < CDIM)  // refill it with tile k+2
            tdm_load_wtile(Wg, k0 + 2 * KSTEP, cur ? LDS_W1 : LDS_W0);
#endif
    }

    // epilogue
    #pragma unroll
    for (int j = 0; j < 8; ++j) {
        const int o = (nb0 + j) * 16 + l15;
        float inv = 1.f, add = 0.f;
        if constexpr (EPI == 1 || EPI == 2) {
            inv = gm[o] * rsqrtf(vr[o] + BN_EPS);
            add = bt[o] - mu[o] * inv;
        }
        #pragma unroll
        for (int v = 0; v < 8; ++v) {
            float x = acc[j][v];
            if constexpr (EPI == 0)             x = fmaxf(x, 0.f);
            if constexpr (EPI == 1 || EPI == 2) x = fmaxf(x * inv + add, 0.f);
            if constexpr (EPI <= 1) {
                const int m = mi * 16 + v + 8 * lhalf;   // C/D row layout
                sDst[m * XSTR + o] = f2bf(x);
            } else {
                acc[j][v] = x;
            }
        }
        if constexpr (EPI >= 2) accOut[j] = acc[j];
    }
}

// One-shot fp32 -> bf16 conversion of the four 512x512 weight matrices.
__global__ __launch_bounds__(256)
void convert_weights_kernel(const float* __restrict__ Wk,
                            const float* __restrict__ Wv1,
                            const float* __restrict__ Wv2,
                            const float* __restrict__ Wvc,
                            unsigned short* __restrict__ dst)
{
    const int idx = blockIdx.x * 256 + threadIdx.x;   // float4 group id
    const int m   = idx >> 16;                        // 65536 groups / matrix
    const int g   = idx & 0xFFFF;
    const float* src = (m == 0) ? Wk : (m == 1) ? Wv1 : (m == 2) ? Wv2 : Wvc;
    f32x4 f = *(const f32x4*)(src + (size_t)g * 4);
    u16x4 o;
    o.x = f2bf(f.x); o.y = f2bf(f.y); o.z = f2bf(f.z); o.w = f2bf(f.w);
    *(u16x4*)(dst + (size_t)idx * 4) = o;
}

__global__ __launch_bounds__(512)
void attn_fused_kernel(const float* __restrict__ features,
                       const float* __restrict__ features2,
                       const unsigned short* __restrict__ Wk,
                       const unsigned short* __restrict__ Wv1,
                       const unsigned short* __restrict__ Wv2,
                       const float* __restrict__ g1, const float* __restrict__ b1,
                       const float* __restrict__ m1, const float* __restrict__ v1,
                       const float* __restrict__ g2, const float* __restrict__ b2,
                       const float* __restrict__ m2, const float* __restrict__ v2,
                       const unsigned short* __restrict__ Wvc,
                       float* __restrict__ out)
{
    extern __shared__ __align__(16) char smem[];
    unsigned short* buf0 = (unsigned short*)(smem + LDS_BUF0);
    unsigned short* buf1 = (unsigned short*)(smem + LDS_BUF1);
    unsigned short* sPos = (unsigned short*)(smem + LDS_POS);
    unsigned short* sW   = (unsigned short*)(smem + LDS_W0);

    const int    tid = threadIdx.x;
    const size_t r0  = (size_t)blockIdx.x * MT;

    // Stage the 64x512 fp32 input slabs as bf16 into LDS (coalesced f32x4).
    #pragma unroll
    for (int i = 0; i < 16; ++i) {
        const int idx = tid + i * 512;           // 0..8191 float4 groups
        const int m   = idx >> 7;                // row within slab
        const int c   = (idx & 127) * 4;         // column
        f32x4 fa = *(const f32x4*)(features  + (r0 + m) * CDIM + c);
        f32x4 fb = *(const f32x4*)(features2 + (r0 + m) * CDIM + c);
        unsigned short* da = buf0 + m * XSTR + c;
        unsigned short* db = buf1 + m * XSTR + c;
        da[0]=f2bf(fa.x); da[1]=f2bf(fa.y); da[2]=f2bf(fa.z); da[3]=f2bf(fa.w);
        db[0]=f2bf(fb.x); db[1]=f2bf(fb.y); db[2]=f2bf(fb.z); db[3]=f2bf(fb.w);
    }
    // (gemm_pass's first barrier orders the stage before any reads)

    // pos = relu(features @ Wk^T)           -> sPos (bf16)
    gemm_pass<0>(buf0, sW, Wk,  sPos, nullptr,
                 nullptr, nullptr, nullptr, nullptr, tid);
    // h = relu(bn1(features2 @ Wv1^T))      -> buf0 (aliases dead features tile)
    gemm_pass<1>(buf1, sW, Wv1, buf0, nullptr, g1, b1, m1, v1, tid);
    // value = relu(bn2(h @ Wv2^T))          -> registers
    v8f valA[8];
    gemm_pass<2>(buf0, sW, Wv2, nullptr, valA, g2, b2, m2, v2, tid);
    // vcl = pos @ Wvc^T (pre-sigmoid)       -> registers
    v8f vcA[8];
    gemm_pass<3>(sPos, sW, Wvc, nullptr, vcA,
                 nullptr, nullptr, nullptr, nullptr, tid);

    // out = value + pos * sigmoid(vcl)   (attention path cancels exactly)
    const int lane  = tid & 31;
    const int wave  = tid >> 5;
    const int mi    = wave & 3;
    const int nb0   = (wave >> 2) * 8;
    const int lhalf = lane >> 4;
    const int l15   = lane & 15;
    #pragma unroll
    for (int j = 0; j < 8; ++j) {
        const int o = (nb0 + j) * 16 + l15;
        #pragma unroll
        for (int v = 0; v < 8; ++v) {
            const int m  = mi * 16 + v + 8 * lhalf;
            const float vc = 1.0f / (1.0f + __expf(-vcA[j][v]));
            const float p  = bf2f(sPos[m * XSTR + o]);
            out[(r0 + m) * DDIM + o] = valA[j][v] + p * vc;
        }
    }
}

extern "C" void kernel_launch(void* const* d_in, const int* in_sizes, int n_in,
                              void* d_out, int out_size, void* d_ws, size_t ws_size,
                              hipStream_t stream) {
    (void)in_sizes; (void)n_in; (void)out_size; (void)ws_size;
    const float* features  = (const float*)d_in[0];
    const float* features2 = (const float*)d_in[1];
    const float* Wk        = (const float*)d_in[2];
    const float* Wv1       = (const float*)d_in[3];
    const float* Wv2       = (const float*)d_in[4];
    const float* g1        = (const float*)d_in[5];
    const float* b1        = (const float*)d_in[6];
    const float* m1        = (const float*)d_in[7];
    const float* v1        = (const float*)d_in[8];
    const float* g2        = (const float*)d_in[9];
    const float* b2        = (const float*)d_in[10];
    const float* m2        = (const float*)d_in[11];
    const float* v2        = (const float*)d_in[12];
    // d_in[13] = Wa, d_in[14] = Wqk : cancelled by the exactly-uniform softmax
    const float* Wvc       = (const float*)d_in[15];
    float* out = (float*)d_out;

    unsigned short* wsb  = (unsigned short*)d_ws;     // 4 x 512KB bf16 weights
    unsigned short* Wkb  = wsb + 0 * WELEMS;
    unsigned short* Wv1b = wsb + 1 * WELEMS;
    unsigned short* Wv2b = wsb + 2 * WELEMS;
    unsigned short* Wvcb = wsb + 3 * WELEMS;

    convert_weights_kernel<<<dim3(1024), dim3(256), 0, stream>>>(
        Wk, Wv1, Wv2, Wvc, wsb);

    attn_fused_kernel<<<dim3(NROIS / MT), dim3(512), LDS_TOTAL, stream>>>(
        features, features2, Wkb, Wv1b, Wv2b,
        g1, b1, m1, v1, g2, b2, m2, v2, Wvcb, out);
}
